// CpGModuleWithWindow_91044716741088
// MI455X (gfx1250) — matrix operand order repaired and checked by reading, not compiled, not run on hardware
//
#include <hip/hip_runtime.h>
#include <hip/hip_bf16.h>

// ---------------------------------------------------------------------------
// Problem constants (from reference)
// ---------------------------------------------------------------------------
#define B_   2
#define S_   48
#define C_   10
#define W_   101
#define D_   256
#define K_   10
#define H_   8
#define DH_  32
#define NWIN (B_ * S_ * C_)   // 960 windows
#define TPAD 112              // padded time axis: [0..4]=0, [5..105]=data, [106..111]=0
#define EPS_ 1e-5f

#define SST  264              // src LDS row stride (halves): 4-bank walk, 16B aligned
#define SROWS 144             // TPAD + tap overhang, rounded up
#define AST  40               // A/B tile LDS row stride (halves): 20-bank walk

typedef __attribute__((ext_vector_type(16))) _Float16 v16h;
typedef __attribute__((ext_vector_type(8)))  _Float16 v8h;
typedef __attribute__((ext_vector_type(4)))  _Float16 v4h;
typedef __attribute__((ext_vector_type(8)))  float    v8f;
typedef __attribute__((ext_vector_type(4)))  int      v4i;

// ---------------------------------------------------------------------------
// Async global->LDS staging (GLOBAL_LOAD_ASYNC_TO_LDS_B128, ASYNCcnt) when the
// toolchain declares the gfx1250 builtins; otherwise fall back to the proven
// b128 load/store path. Signature (from the round-4 diagnostic): 4 args,
// param1 = int4* in global AS, param2 = int4* in LDS AS, then imm offset/cpol.
// ---------------------------------------------------------------------------
#if __has_builtin(__builtin_amdgcn_global_load_async_to_lds_b128)
#define HAS_ASYNC_LDS 1
#else
#define HAS_ASYNC_LDS 0
#endif

#if HAS_ASYNC_LDS
typedef __attribute__((address_space(1))) v4i* gas_ptr;
typedef __attribute__((address_space(3))) v4i* las_ptr;
__device__ __forceinline__ void async_wait0() {
#if __has_builtin(__builtin_amdgcn_s_wait_asynccnt)
  __builtin_amdgcn_s_wait_asynccnt(0);
#else
  asm volatile("s_wait_asynccnt 0x0" ::: "memory");
#endif
}
#endif

// ---------------------------------------------------------------------------
// WMMA fragment loads from LDS (gfx1250 wave32 layouts), all 2x ds_load_b128:
//   A (16x32 f16): lane m=l&15; halves [kb,kb+8) and [16+kb,16+kb+8), kb=(l>>4)*8
//   B (32x16 f16): lane n=l&15; halves [kb,kb+8) and [16+kb,16+kb+8), kb=(l>>4)*8
//   C/D (16x16 f32): VGPR r -> row r (lanes 0-15) / 8+r (lanes 16-31), col l&15
// ---------------------------------------------------------------------------
__device__ __forceinline__ v16h frag_a(const _Float16* tile, int stride, int m0, int lane) {
  const _Float16* rp = tile + (m0 + (lane & 15)) * stride + ((lane >> 4) << 3);
  v8h lo = *(const v8h*)rp;
  v8h hi = *(const v8h*)(rp + 16);
  v16h r;
#pragma unroll
  for (int i = 0; i < 8; ++i) { r[i] = lo[i]; r[i + 8] = hi[i]; }
  return r;
}

__device__ __forceinline__ v16h frag_b(const _Float16* tile, int stride, int n0, int lane) {
  const _Float16* rp = tile + (n0 + (lane & 15)) * stride + ((lane >> 4) << 3);
  v8h lo = *(const v8h*)rp;
  v8h hi = *(const v8h*)(rp + 16);
  v16h r;
#pragma unroll
  for (int i = 0; i < 8; ++i) { r[i] = lo[i]; r[i + 8] = hi[i]; }
  return r;
}

// ---------------------------------------------------------------------------
// Kernel 0: one-time conv weight transform: w[oc][ic][kt] f32 -> wt[kt][oc][ic]
// f16 (tap-major, ic-contiguous => v8h-loadable A tiles; L2-resident, 1.3MB).
// ---------------------------------------------------------------------------
__global__ __launch_bounds__(256) void cvt_wt_kernel(
    const float* __restrict__ w, _Float16* __restrict__ wt) {
  int idx = blockIdx.x * 256 + threadIdx.x;   // (oc,ic) pair, 65536 total
  int oc = idx >> 8, ic = idx & 255;
#pragma unroll
  for (int kt = 0; kt < K_; ++kt)
    wt[((size_t)kt * D_ + oc) * D_ + ic] = (_Float16)w[(size_t)idx * K_ + kt];
}

// ---------------------------------------------------------------------------
// Kernel 1: window extraction + conv1 (1->256) + BN fold + ReLU.
// Output layout TRANSPOSED: dst[win][t][d] (t has zeroed conv-pad halo).
// Stores are lane-coalesced (256 contiguous halves per t).
// ---------------------------------------------------------------------------
__global__ __launch_bounds__(256) void conv1_kernel(
    const float* __restrict__ x,   const float* __restrict__ w0,
    const float* __restrict__ b0,  const float* __restrict__ bn_s,
    const float* __restrict__ bn_b,const float* __restrict__ bn_m,
    const float* __restrict__ bn_v,_Float16* __restrict__ dst) {
  __shared__ float wpad[TPAD];
  const int win = blockIdx.x;
  const int c   = win % C_;
  const int bs  = win / C_;
  const int s   = bs % S_;
  const int b   = bs / S_;
  const int tid = threadIdx.x;

  if (tid < TPAD) {
    float v = 0.f;                 // conv zero-padding
    int j = tid - 5;               // window position 0..100
    if (j >= 0 && j < W_) {
      int srow = s + j - (W_ / 2);
      v = (srow >= 0 && srow < S_) ? x[(b * S_ + srow) * C_ + c] : -1.0f;
    }
    wpad[tid] = v;
  }
  __syncthreads();

  const int d = tid;
  float wk[K_];
#pragma unroll
  for (int k = 0; k < K_; ++k) wk[k] = w0[d * K_ + k];
  const float g    = bn_s[d] * rsqrtf(bn_v[d] + EPS_);
  const float bias = b0[d] * g + bn_b[d] - bn_m[d] * g;

  _Float16* out = dst + (size_t)win * TPAD * D_ + d;
#pragma unroll
  for (int p = 0; p < 5; ++p)      out[(size_t)p * D_] = (_Float16)0.f;
#pragma unroll
  for (int p = 106; p < TPAD; ++p) out[(size_t)p * D_] = (_Float16)0.f;
  for (int t = 0; t < W_; ++t) {
    float acc = 0.f;
#pragma unroll
    for (int k = 0; k < K_; ++k) acc += wk[k] * wpad[t + k];
    out[(size_t)(t + 5) * D_] = (_Float16)fmaxf(acc * g + bias, 0.f);
  }
}

// ---------------------------------------------------------------------------
// Kernel 2: conv2/conv3 as tap-decomposed implicit GEMM on WMMA.
//   out[win][p][oc] = relu(BN( sum_kt sum_ic wt[kt][oc][ic] * in[win][p+kt][ic] ))
// Block = 256 threads (8 waves). Tile: M=64 oc, N=128 p (112 valid, 8 n-tiles).
// Wave w: m-tile (w&3), n-tiles (w>>2)*4..+3 => 4 WMMA sharing one A fragment.
// K decomposed as kt(10 taps) x ic-chunks(8 x 32): with the [t][ic] layout the
// B fragment is a DIRECT wide LDS load from the staged input panel (no im2col
// staging, no integer division). 40 WMMA per wave between barrier pairs.
// Staging uses async global->LDS DMA when available (ASYNCcnt path).
// ---------------------------------------------------------------------------
__global__ __launch_bounds__(256) void conv_gemm_kernel(
    const _Float16* __restrict__ srcT,  // [win][TPAD][256]
    const _Float16* __restrict__ wt,    // [kt][oc][ic] f16
    const float* __restrict__ bconv,  const float* __restrict__ bn_s,
    const float* __restrict__ bn_b,   const float* __restrict__ bn_m,
    const float* __restrict__ bn_v,   _Float16* __restrict__ dstT) {
  __shared__ __align__(16) _Float16 s_in[SROWS * SST];     // 76,032 B
  __shared__ __align__(16) _Float16 s_w[K_ * 64 * AST];    // 51,200 B

  const int win  = blockIdx.y;
  const int oc0  = blockIdx.x * 64;
  const int tid  = threadIdx.x;
  const int lane = tid & 31;
  const int wv   = tid >> 5;
  const int wm   = (wv & 3) * 16;       // m-tile within 64
  const int wn0  = (wv >> 2) * 64;      // first of 4 n-tiles

  // Stage window input (rows 0..111 data, 112..143 zero).
  {
    const _Float16* gsrc = srcT + (size_t)win * TPAD * D_;
#pragma unroll
    for (int it = 0; it < 14; ++it) {              // 112 rows * 32 v8h / 256
      int gi = it * 256 + tid;
      int rr = gi >> 5, c8 = (gi & 31) << 3;
#if HAS_ASYNC_LDS
      __builtin_amdgcn_global_load_async_to_lds_b128(
          (gas_ptr)(gsrc + (size_t)gi * 8), (las_ptr)(s_in + rr * SST + c8), 0, 0);
#else
      *(v8h*)(s_in + rr * SST + c8) = *(const v8h*)(gsrc + (size_t)gi * 8);
#endif
    }
    v8h z = {};
#pragma unroll
    for (int it = 0; it < 4; ++it) {               // 32 pad rows * 32 / 256
      int gi = it * 256 + tid;
      int rr = 112 + (gi >> 5), c8 = (gi & 31) << 3;
      *(v8h*)(s_in + rr * SST + c8) = z;
    }
  }

  v8f acc[4] = {};
  for (int ic0 = 0; ic0 < D_; ic0 += 32) {
#if HAS_ASYNC_LDS
    async_wait0();                                  // our async stores visible
#endif
    __syncthreads();
    // Stage all 10 taps' A tiles: wt[kt][oc0+m][ic0..ic0+32)
#pragma unroll
    for (int it = 0; it < 10; ++it) {              // 10*64*4 groups / 256
      int gi = it * 256 + tid;
      int kt = gi >> 8;
      int m  = (gi >> 2) & 63;
      int k8 = (gi & 3) << 3;
#if HAS_ASYNC_LDS
      __builtin_amdgcn_global_load_async_to_lds_b128(
          (gas_ptr)(wt + ((size_t)kt * D_ + oc0 + m) * D_ + ic0 + k8),
          (las_ptr)(s_w + (kt * 64 + m) * AST + k8), 0, 0);
#else
      *(v8h*)(s_w + (kt * 64 + m) * AST + k8) =
          *(const v8h*)(wt + ((size_t)kt * D_ + oc0 + m) * D_ + ic0 + k8);
#endif
    }
#if HAS_ASYNC_LDS
    async_wait0();
#endif
    __syncthreads();

#pragma unroll
    for (int kt = 0; kt < K_; ++kt) {
      v16h a = frag_a(s_w + kt * 64 * AST, AST, wm, lane);
#pragma unroll
      for (int j = 0; j < 4; ++j) {
        // B element (ka, col p) = s_in[(p+kt)*SST + ic0+ka]; p-5+kt = time idx
        v16h b = frag_b(s_in + kt * SST + ic0, SST, wn0 + j * 16, lane);
        acc[j] = __builtin_amdgcn_wmma_f32_16x16x32_f16(false, a, false, b,
                                                        (short)0, acc[j], false, false);
      }
    }
  }

  // Epilogue: fused bias+BN+ReLU; 8 consecutive oc per lane => one b128 store.
  const int mb = oc0 + wm + ((lane >> 4) << 3);
  float garr[8], barr[8];
#pragma unroll
  for (int r = 0; r < 8; ++r) {
    float gg = bn_s[mb + r] * rsqrtf(bn_v[mb + r] + EPS_);
    garr[r] = gg;
    barr[r] = bconv[mb + r] * gg + bn_b[mb + r] - bn_m[mb + r] * gg;
  }
  _Float16* dbase = dstT + (size_t)win * TPAD * D_;
#pragma unroll
  for (int j = 0; j < 4; ++j) {
    int p = wn0 + j * 16 + (lane & 15);
    if (p < TPAD) {
      bool valid = (p >= 5 && p <= 105);   // zero the conv-pad halo
      v8h o;
#pragma unroll
      for (int r = 0; r < 8; ++r) {
        float v = valid ? fmaxf(acc[j][r] * garr[r] + barr[r], 0.f) : 0.f;
        o[r] = (_Float16)v;
      }
      *(v8h*)(dbase + (size_t)p * D_ + mb) = o;
    }
  }
}

// ---------------------------------------------------------------------------
// Kernel 3/4: feature reductions (transposed [win][t][d] layout, coalesced).
// ---------------------------------------------------------------------------
__global__ __launch_bounds__(256) void gather_cl_kernel(
    const _Float16* __restrict__ z3, float* __restrict__ cl) {
  const int win = blockIdx.x, d = threadIdx.x;
  cl[(size_t)win * D_ + d] =
      (float)z3[(size_t)win * TPAD * D_ + (size_t)(W_ / 2 + 5) * D_ + d];
}

__global__ __launch_bounds__(256) void reduce_pos_kernel(
    const _Float16* __restrict__ z3, float* __restrict__ pos) {
  const int t  = blockIdx.x % W_;
  const int bs = blockIdx.x / W_;
  const int d  = threadIdx.x;
  float s = 0.f;
  for (int c = 0; c < C_; ++c)
    s += (float)z3[(size_t)(bs * C_ + c) * TPAD * D_ + (size_t)(t + 5) * D_ + d];
  pos[((size_t)bs * W_ + t) * D_ + d] = s * (1.0f / C_);
}

// ---------------------------------------------------------------------------
// Kernel 5: generic WMMA GEMM  C[M,N] = act(A[M,K] @ Bw[N,K]^T + bias[N]).
// Block = 256 threads (8 waves), tile 64(M) x 64(N); wave: 2 WMMA, shared A.
// f32->f16 in flight (float4 -> v4h), padded LDS stride AST (bank-spread).
// ---------------------------------------------------------------------------
__global__ __launch_bounds__(256) void gemm_kernel(
    const float* __restrict__ A, const float* __restrict__ Bw,
    const float* __restrict__ bias, float* __restrict__ Cm,
    int M, int N, int Kd, int act) {
  __shared__ __align__(16) _Float16 As[64 * AST];
  __shared__ __align__(16) _Float16 Bs[64 * AST];   // [n][k] (k-contiguous)

  const int m0 = blockIdx.y * 64;
  const int n0 = blockIdx.x * 64;
  const int tid  = threadIdx.x;
  const int lane = tid & 31;
  const int wv   = tid >> 5;
  const int wm   = (wv & 3) * 16;
  const int wn   = (wv >> 2) * 32;

  v8f acc0 = {}, acc1 = {};
  for (int k0 = 0; k0 < Kd; k0 += 32) {
#pragma unroll
    for (int it = 0; it < 2; ++it) {             // A: 64x32 / 4 elems
      int g = it * 256 + tid;
      int m = g >> 3, k4 = (g & 7) << 2;
      int gm = m0 + m;
      float4 v = make_float4(0.f, 0.f, 0.f, 0.f);
      if (gm < M) v = *(const float4*)(A + (size_t)gm * Kd + k0 + k4);
      v4h h = {(_Float16)v.x, (_Float16)v.y, (_Float16)v.z, (_Float16)v.w};
      *(v4h*)(As + m * AST + k4) = h;
    }
#pragma unroll
    for (int it = 0; it < 2; ++it) {             // B: 64x32 / 4 elems
      int g = it * 256 + tid;
      int n = g >> 3, k4 = (g & 7) << 2;
      int gn = n0 + n;
      float4 v = make_float4(0.f, 0.f, 0.f, 0.f);
      if (gn < N) v = *(const float4*)(Bw + (size_t)gn * Kd + k0 + k4);
      v4h h = {(_Float16)v.x, (_Float16)v.y, (_Float16)v.z, (_Float16)v.w};
      *(v4h*)(Bs + n * AST + k4) = h;
    }
    if (k0 + 32 < Kd) {                          // global_prefetch_b8 next chunk
      __builtin_prefetch(&A[(size_t)(m0 + (tid >> 3)) * Kd + k0 + 32], 0, 1);
      __builtin_prefetch(&Bw[(size_t)(n0 + (tid >> 3)) * Kd + k0 + 32], 0, 1);
    }
    __syncthreads();

    v16h a  = frag_a(As, AST, wm, lane);
    v16h b0 = frag_b(Bs, AST, wn, lane);
    v16h b1 = frag_b(Bs, AST, wn + 16, lane);
    acc0 = __builtin_amdgcn_wmma_f32_16x16x32_f16(false, a, false, b0,
                                                  (short)0, acc0, false, false);
    acc1 = __builtin_amdgcn_wmma_f32_16x16x32_f16(false, a, false, b1,
                                                  (short)0, acc1, false, false);
    __syncthreads();
  }

  const int gmb = m0 + wm + ((lane >> 4) << 3);
#pragma unroll
  for (int half = 0; half < 2; ++half) {
    const v8f& acc = half ? acc1 : acc0;
    int gn = n0 + wn + half * 16 + (lane & 15);
    float bb = (gn < N) ? bias[gn] : 0.f;
#pragma unroll
    for (int r = 0; r < 8; ++r) {
      int gm = gmb + r;
      if (gm < M && gn < N) {
        float v = acc[r] + bb;
        if (act == 1) v = fmaxf(v, 0.f);
        Cm[(size_t)gm * N + gn] = v;
      }
    }
  }
}

// ---------------------------------------------------------------------------
// Kernel 6: fused softmax attention, one (token-batch n, head h) per block.
// qkv layout: [(n*T + t)*768 + {0|256|512} + h*32 + d]. T <= 101.
// VALU path (attention is ~2% of total FLOPs).
// ---------------------------------------------------------------------------
__global__ __launch_bounds__(128) void attn_kernel(
    const float* __restrict__ qkv, float* __restrict__ obuf, int T) {
  __shared__ float ks[W_ * DH_];
  __shared__ float vs[W_ * DH_];
  __shared__ float qs[DH_];
  __shared__ float sc[W_ + 1];

  const int h = blockIdx.x % H_;
  const int n = blockIdx.x / H_;
  const int tid = threadIdx.x;
  const float* base = qkv + (size_t)n * T * 768;

  for (int idx = tid; idx < T * DH_; idx += 128) {
    int t = idx / DH_, d = idx % DH_;
    ks[idx] = base[(size_t)t * 768 + 256 + h * DH_ + d];
    vs[idx] = base[(size_t)t * 768 + 512 + h * DH_ + d];
  }
  __syncthreads();

  const float scale = 0.17677669529663687f; // 1/sqrt(32)
  for (int i = 0; i < T; ++i) {
    if (tid < DH_) qs[tid] = base[(size_t)i * 768 + h * DH_ + tid];
    __syncthreads();
    for (int j = tid; j < T; j += 128) {
      float dsum = 0.f;
#pragma unroll
      for (int d = 0; d < DH_; ++d) dsum += qs[d] * ks[j * DH_ + d];
      sc[j] = dsum * scale;
    }
    __syncthreads();
    if (tid == 0) {
      float mx = sc[0];
      for (int j = 1; j < T; ++j) mx = fmaxf(mx, sc[j]);
      float ssum = 0.f;
      for (int j = 0; j < T; ++j) { float e = __expf(sc[j] - mx); sc[j] = e; ssum += e; }
      sc[W_] = 1.f / ssum;
    }
    __syncthreads();
    if (tid < DH_) {
      float inv = sc[W_], o = 0.f;
      for (int j = 0; j < T; ++j) o += sc[j] * vs[j * DH_ + tid];
      obuf[((size_t)n * T + i) * D_ + h * DH_ + tid] = o * inv;
    }
    __syncthreads();
  }
}

// ---------------------------------------------------------------------------
// Kernel 7: fused residual-add + LayerNorm over d=256, in place on z.
// ---------------------------------------------------------------------------
__global__ __launch_bounds__(256) void resln_kernel(
    float* __restrict__ z, const float* __restrict__ t,
    const float* __restrict__ s, const float* __restrict__ b) {
  __shared__ float red[8];
  const int row = blockIdx.x, d = threadIdx.x;
  float v = z[(size_t)row * D_ + d] + t[(size_t)row * D_ + d];

  float x = v;
#pragma unroll
  for (int o = 16; o > 0; o >>= 1) x += __shfl_down(x, o, 32);
  if ((d & 31) == 0) red[d >> 5] = x;
  __syncthreads();
  if (d == 0) {
    float sm = 0.f;
    for (int i = 0; i < 8; ++i) sm += red[i];
    red[0] = sm / D_;
  }
  __syncthreads();
  const float mean = red[0];
  __syncthreads();

  const float dv = v - mean;
  float xx = dv * dv;
#pragma unroll
  for (int o = 16; o > 0; o >>= 1) xx += __shfl_down(xx, o, 32);
  if ((d & 31) == 0) red[d >> 5] = xx;
  __syncthreads();
  if (d == 0) {
    float sm = 0.f;
    for (int i = 0; i < 8; ++i) sm += red[i];
    red[0] = sm / D_;
  }
  __syncthreads();
  z[(size_t)row * D_ + d] = dv * rsqrtf(red[0] + EPS_) * s[d] + b[d];
}

// ---------------------------------------------------------------------------
// Kernel 8/9: final heads.
// ---------------------------------------------------------------------------
__global__ __launch_bounds__(256) void cl_mean_kernel(
    const float* __restrict__ zcl, float* __restrict__ out) {
  const int bs = blockIdx.x, d = threadIdx.x;
  float s = 0.f;
  for (int c = 0; c < C_; ++c) s += zcl[((size_t)bs * C_ + c) * D_ + d];
  out[(size_t)bs * D_ + d] = s * (1.0f / C_);
}

__global__ __launch_bounds__(256) void pos_pick_kernel(
    const float* __restrict__ zpos, float* __restrict__ out) {
  const int bs = blockIdx.x, d = threadIdx.x;
  out[(size_t)bs * D_ + d] = zpos[((size_t)bs * W_ + (W_ / 2)) * D_ + d];
}

// ---------------------------------------------------------------------------
// Host side
// ---------------------------------------------------------------------------
struct EncP {
  const float *wqkv, *bqkv, *wo, *bo, *ln1s, *ln1b, *w1, *b1, *w2, *b2, *ln2s, *ln2b;
};

static inline EncP mk_enc(void* const* d_in, int base) {
  EncP p;
  p.wqkv = (const float*)d_in[base + 0];  p.bqkv = (const float*)d_in[base + 1];
  p.wo   = (const float*)d_in[base + 2];  p.bo   = (const float*)d_in[base + 3];
  p.ln1s = (const float*)d_in[base + 4];  p.ln1b = (const float*)d_in[base + 5];
  p.w1   = (const float*)d_in[base + 6];  p.b1   = (const float*)d_in[base + 7];
  p.w2   = (const float*)d_in[base + 8];  p.b2   = (const float*)d_in[base + 9];
  p.ln2s = (const float*)d_in[base + 10]; p.ln2b = (const float*)d_in[base + 11];
  return p;
}

static void run_encoder(float* z, int T, const EncP& p,
                        float* qkvb, float* tmpb, float* hb, hipStream_t stream) {
  const int M = (B_ * S_) * T;
  const unsigned gy = (unsigned)((M + 63) / 64);
  const dim3 gm(768 / 64, gy);
  const dim3 go(256 / 64, gy);
  const dim3 gf(1024 / 64, gy);
  for (int i = 0; i < 8; ++i) {
    gemm_kernel<<<gm, 256, 0, stream>>>(z, p.wqkv + (size_t)i * 768 * 256,
                                        p.bqkv + (size_t)i * 768, qkvb, M, 768, 256, 0);
    attn_kernel<<<(B_ * S_) * H_, 128, 0, stream>>>(qkvb, hb, T);
    gemm_kernel<<<go, 256, 0, stream>>>(hb, p.wo + (size_t)i * 256 * 256,
                                        p.bo + (size_t)i * 256, tmpb, M, 256, 256, 0);
    resln_kernel<<<M, 256, 0, stream>>>(z, tmpb, p.ln1s + (size_t)i * 256,
                                        p.ln1b + (size_t)i * 256);
    gemm_kernel<<<gf, 256, 0, stream>>>(z, p.w1 + (size_t)i * 1024 * 256,
                                        p.b1 + (size_t)i * 1024, hb, M, 1024, 256, 1);
    gemm_kernel<<<go, 256, 0, stream>>>(hb, p.w2 + (size_t)i * 256 * 1024,
                                        p.b2 + (size_t)i * 256, tmpb, M, 256, 1024, 0);
    resln_kernel<<<M, 256, 0, stream>>>(z, tmpb, p.ln2s + (size_t)i * 256,
                                        p.ln2b + (size_t)i * 256);
  }
}

extern "C" void kernel_launch(void* const* d_in, const int* in_sizes, int n_in,
                              void* d_out, int out_size, void* d_ws, size_t ws_size,
                              hipStream_t stream) {
  (void)in_sizes; (void)n_in; (void)out_size; (void)ws_size;
  // Input order: x, conv{w0,b0,w1,b1,w2,b2,bn_s,bn_b,bn_m,bn_v}, cl{12}, pos{12}
  const float* x    = (const float*)d_in[0];
  const float* w0   = (const float*)d_in[1];
  const float* b0   = (const float*)d_in[2];
  const float* w1c  = (const float*)d_in[3];
  const float* b1c  = (const float*)d_in[4];
  const float* w2c  = (const float*)d_in[5];
  const float* b2c  = (const float*)d_in[6];
  const float* bn_s = (const float*)d_in[7];
  const float* bn_b = (const float*)d_in[8];
  const float* bn_m = (const float*)d_in[9];
  const float* bn_v = (const float*)d_in[10];
  const EncP clp  = mk_enc(d_in, 11);
  const EncP posp = mk_enc(d_in, 23);

  // Workspace layout (bytes, all offsets 256B aligned)
  const size_t ZSZ = (size_t)NWIN * TPAD * D_ * sizeof(_Float16);   // 55,050,240
  char* ws = (char*)d_ws;
  _Float16* zA  = (_Float16*)(ws);
  _Float16* zB  = (_Float16*)(ws + ZSZ);
  float*    zcl = (float*)(ws + 2 * ZSZ);                            // 96*10*256 f32
  float*    zpo = (float*)(ws + 2 * ZSZ + (size_t)96 * C_ * D_ * 4); // 96*101*256 f32
  _Float16* wt1 = (_Float16*)(ws + 121012224);                       // 10*256*256 f16
  _Float16* wt2 = (_Float16*)(ws + 122322944);                       // 10*256*256 f16
  // Encoder scratch overlaps the (dead-by-then) conv ping-pong buffers:
  float* qkvb = (float*)(ws);                                        // 96*101*768 f32
  float* tmpb = (float*)(ws + 29786112);                             // 96*101*256 f32
  float* hb   = (float*)(ws + 39714816);                             // 96*101*1024 f32

  // One-time weight transform (f16 tap-major; L2-resident for all conv blocks)
  cvt_wt_kernel<<<D_ * D_ / 256, 256, 0, stream>>>(w1c, wt1);
  cvt_wt_kernel<<<D_ * D_ / 256, 256, 0, stream>>>(w2c, wt2);

  // Conv stack (WMMA implicit GEMM, f16 activations, fused BN+ReLU)
  conv1_kernel<<<NWIN, 256, 0, stream>>>(x, w0, b0, bn_s, bn_b, bn_m, bn_v, zA);
  conv_gemm_kernel<<<dim3(D_ / 64, NWIN), 256, 0, stream>>>(
      zA, wt1, b1c, bn_s + 256, bn_b + 256, bn_m + 256, bn_v + 256, zB);
  conv_gemm_kernel<<<dim3(D_ / 64, NWIN), 256, 0, stream>>>(
      zB, wt2, b2c, bn_s + 512, bn_b + 512, bn_m + 512, bn_v + 512, zA);

  // Reductions into persistent encoder token buffers
  gather_cl_kernel<<<NWIN, 256, 0, stream>>>(zA, zcl);
  reduce_pos_kernel<<<(B_ * S_) * W_, 256, 0, stream>>>(zA, zpo);

  // Encoders (conv buffers now reusable as scratch)
  run_encoder(zcl, C_, clp, qkvb, tmpb, hb, stream);
  cl_mean_kernel<<<B_ * S_, 256, 0, stream>>>(zcl, (float*)d_out);

  run_encoder(zpo, W_, posp, qkvb, tmpb, hb, stream);
  pos_pick_kernel<<<B_ * S_, 256, 0, stream>>>(zpo, (float*)d_out + (size_t)B_ * S_ * D_);
}
